// GPT_16312285790352
// MI455X (gfx1250) — matrix operand order repaired and checked by reading, not compiled, not run on hardware
//
#include <hip/hip_runtime.h>
#include <math.h>

// ---------------- model constants ----------------
constexpr int CB = 2;      // batch
constexpr int CT = 1024;   // seq len
constexpr int CE = 768;    // embed
constexpr int CH = 12;     // heads
constexpr int CD = 64;     // head dim
constexpr int CV = 50257;  // vocab
constexpr float CSCALE = 2660.0f;  // faithful quirk: E // (H**-0.5)
constexpr float CEPS = 1e-5f;

// ---------------- WMMA types (gfx1250, wave32) ----------------
typedef __bf16 v16bf __attribute__((ext_vector_type(16)));
typedef __bf16 v8bf  __attribute__((ext_vector_type(8)));
typedef float  v8f   __attribute__((ext_vector_type(8)));

union FragBF {
  v16bf v;
  v8bf h[2];
  unsigned int u[8];
  unsigned short s[16];
};

__device__ inline v8f zero8() { v8f z = {}; return z; }

// fp32 -> bf16 round-to-nearest-even (bit manip; no dependence on __bf16 arith)
__device__ inline unsigned short f2bf(float f) {
  unsigned int u = __float_as_uint(f);
  return (unsigned short)((u + 0x7fffu + ((u >> 16) & 1u)) >> 16);
}

__device__ inline float gelu_exact(float x) {
  return 0.5f * x * (1.0f + erff(x * 0.70710678118654752f));
}

// ---- CDNA5 async-to-LDS helpers ----
template <int N>
__device__ inline void async_wait() {
#if __has_builtin(__builtin_amdgcn_s_wait_asynccnt)
  __builtin_amdgcn_s_wait_asynccnt(N);  // immediate required
#else
  asm volatile("s_wait_asynccnt %0" :: "i"(N) : "memory");
#endif
}

// One b128 async copy per lane: LDS[ldsaddr] = MEM[gaddr], tracked by ASYNCcnt.
__device__ inline void async_b128_to_lds(unsigned ldsaddr, const void* gaddr) {
  asm volatile("global_load_async_to_lds_b128 %0, %1, off"
               :: "v"(ldsaddr), "v"((unsigned long long)(size_t)gaddr)
               : "memory");
}

// ---------------- embedding: x = wte[tok] + wpe ----------------
__global__ void k_embed(const int* __restrict__ tok, const float* __restrict__ wte,
                        const float* __restrict__ wpe, float* __restrict__ x) {
  int idx = blockIdx.x * blockDim.x + threadIdx.x;
  if (idx >= CB * CT * CE) return;
  int e = idx % CE;
  int bt = idx / CE;
  int t = bt % CT;
  x[idx] = wte[(size_t)tok[bt] * CE + e] + wpe[(size_t)t * CE + e];
}

// ---------------- LayerNorm (gamma only) -> bf16 ----------------
__global__ __launch_bounds__(256) void k_ln(const float* __restrict__ x,
                                            const float* __restrict__ g,
                                            unsigned short* __restrict__ out) {
  __shared__ float r1[256], r2[256];
  int row = blockIdx.x;
  const float* px = x + (size_t)row * CE;
  float s = 0.f, ss = 0.f;
  for (int e = threadIdx.x; e < CE; e += 256) { float v = px[e]; s += v; ss += v * v; }
  r1[threadIdx.x] = s; r2[threadIdx.x] = ss;
  __syncthreads();
  for (int st = 128; st > 0; st >>= 1) {
    if ((int)threadIdx.x < st) { r1[threadIdx.x] += r1[threadIdx.x + st]; r2[threadIdx.x] += r2[threadIdx.x + st]; }
    __syncthreads();
  }
  float mu = r1[0] * (1.0f / CE);
  float var = r2[0] * (1.0f / CE) - mu * mu;
  float rinv = rsqrtf(var + CEPS);
  for (int e = threadIdx.x; e < CE; e += 256)
    out[(size_t)row * CE + e] = f2bf((px[e] - mu) * rinv * g[e]);
}

// ---------------- weight transpose+convert: Wt[n,k] = bf16(W[k,n]) ----------------
__global__ void k_transw(const float* __restrict__ W, unsigned short* __restrict__ Wt,
                         int K, int N) {
  size_t idx = (size_t)blockIdx.x * blockDim.x + threadIdx.x;
  if (idx >= (size_t)K * N) return;
  int k = (int)(idx % K);
  int n = (int)(idx / K);
  Wt[idx] = f2bf(W[(size_t)k * N + n]);
}

// ---------------- fp32 -> bf16 elementwise ----------------
__global__ void k_cvt(const float* __restrict__ in, unsigned short* __restrict__ out, int n) {
  int idx = blockIdx.x * blockDim.x + threadIdx.x;
  if (idx < n) out[idx] = f2bf(in[idx]);
}

// ---------------- GEMM: C[M,N] = act(A[M,K]bf16 x Bt[N,K]bf16 + bias) + resid ------
// Block = 8 waves = 32(M) x 512(N) macro-tile. The 32-row A panel is staged to LDS
// with double-buffered global_load_async_to_lds_b128 (ASYNCcnt), shared by all waves.
// Each wave computes 32x64 via 8 WMMA accumulators; B fragments are contiguous b128
// loads from the pre-transposed bf16 weight, with prefetch of the next K-chunk.
__global__ __launch_bounds__(256) void k_gemm(const unsigned short* __restrict__ A,
                                              const unsigned short* __restrict__ Bt,
                                              const float* __restrict__ bias,
                                              const float* __restrict__ resid,
                                              float* __restrict__ C,
                                              int M, int N, int K, int act) {
  __shared__ __align__(16) unsigned short smA[2][32][64];  // 2 x 4KB K-chunks
  const int tid = threadIdx.x;
  const int lane = tid & 31;
  const int wv = tid >> 5;  // 0..7
  const int nsuper = (N + 511) >> 9;
  const int mb = blockIdx.x / nsuper;
  const int ns = blockIdx.x % nsuper;
  const int m0 = mb * 32;
  const int half = lane >> 4, l16 = lane & 15;
  (void)M;

  // B row pointers for this wave's 4 column subtiles (clamped if OOB, masked later)
  const __bf16* brow[4];
#pragma unroll
  for (int j = 0; j < 4; ++j) {
    int n = ns * 512 + wv * 64 + j * 16 + l16;
    brow[j] = (const __bf16*)Bt + (size_t)((n < N) ? n : 0) * K;
  }

  // async staging: thread copies one b128: row = tid>>3 (0..31), seg = tid&7 (0..7)
  const int sr = tid >> 3, sseg = tid & 7;
  const __bf16* asrc = (const __bf16*)A + (size_t)(m0 + sr) * K + sseg * 8;

  v8f acc[8];
#pragma unroll
  for (int j = 0; j < 8; ++j) acc[j] = zero8();

  const int nchunk = K >> 6;  // K is a multiple of 64 here (768 or 3072)
  {
    unsigned dst = (unsigned)(unsigned long long)&smA[0][sr][sseg * 8];
    async_b128_to_lds(dst, asrc);
  }
  for (int c = 0; c < nchunk; ++c) {
    const int buf = c & 1;
    const int kg = c << 6;
    if (c + 1 < nchunk) {
      unsigned dst = (unsigned)(unsigned long long)&smA[buf ^ 1][sr][sseg * 8];
      async_b128_to_lds(dst, asrc + (kg + 64));
#pragma unroll
      for (int j = 0; j < 4; ++j) __builtin_prefetch(brow[j] + kg + 64 + 16 * half, 0, 1);
      async_wait<1>();  // current chunk complete; next still in flight
    } else {
      async_wait<0>();
    }
    __syncthreads();  // all waves' stages visible

#pragma unroll
    for (int kk = 0; kk < 64; kk += 32) {
      FragBF a0, a1;
      // A layout: per-lane K chunks at kk+8*half and kk+16+8*half (ISA 16-bit A 16x32)
      a0.h[0] = *(const v8bf*)&smA[buf][l16][kk + 8 * half];
      a0.h[1] = *(const v8bf*)&smA[buf][l16][kk + 16 + 8 * half];
      a1.h[0] = *(const v8bf*)&smA[buf][16 + l16][kk + 8 * half];
      a1.h[1] = *(const v8bf*)&smA[buf][16 + l16][kk + 16 + 8 * half];
#pragma unroll
      for (int j = 0; j < 4; ++j) {
        FragBF bfj;
        // B layout: per-lane K chunks at kk+8*half and kk+16+8*half (matches A)
        bfj.h[0] = *(const v8bf*)(brow[j] + kg + kk + 8 * half);
        bfj.h[1] = *(const v8bf*)(brow[j] + kg + kk + 16 + 8 * half);
        acc[j] = __builtin_amdgcn_wmma_f32_16x16x32_bf16(
            false, a0.v, false, bfj.v, (short)0, acc[j], false, false);
        acc[4 + j] = __builtin_amdgcn_wmma_f32_16x16x32_bf16(
            false, a1.v, false, bfj.v, (short)0, acc[4 + j], false, false);
      }
    }
    __syncthreads();  // all waves done reading before buffer is overwritten
  }

#pragma unroll
  for (int g = 0; g < 2; ++g) {
#pragma unroll
    for (int j = 0; j < 4; ++j) {
      int n = ns * 512 + wv * 64 + j * 16 + l16;
      if (n >= N) continue;
      float bv = bias ? bias[n] : 0.0f;
#pragma unroll
      for (int r = 0; r < 8; ++r) {
        int row = m0 + g * 16 + r + 8 * half;  // C/D layout: row = r + 8*half
        float v = acc[g * 4 + j][r] + bv;
        if (act == 1) v = gelu_exact(v);
        size_t o = (size_t)row * N + n;
        if (resid) v += resid[o];
        C[o] = v;
      }
    }
  }
}

// ---------------- split qkv into q[B,H,T,D], k[B,H,T,D], v^T[B,H,D,T] (bf16) -------
__global__ void k_split(const float* __restrict__ qkv, unsigned short* __restrict__ qb,
                        unsigned short* __restrict__ kb, unsigned short* __restrict__ vt) {
  int idx = blockIdx.x * blockDim.x + threadIdx.x;
  if (idx >= CB * CT * CE) return;
  int d = idx % CD;
  int tmp = idx / CD;
  int h = tmp % CH; tmp /= CH;
  int t = tmp % CT;
  int b = tmp / CT;
  size_t rb = ((size_t)b * CT + t) * (3 * CE);
  int col = h * CD + d;
  size_t bh = (size_t)(b * CH + h);
  qb[(bh * CT + t) * CD + d] = f2bf(qkv[rb + col]);
  kb[(bh * CT + t) * CD + d] = f2bf(qkv[rb + CE + col]);
  vt[(bh * CD + d) * CT + t] = f2bf(qkv[rb + 2 * CE + col]);
}

// ---------------- scores = scale * Q K^T with causal mask (lower tiles only) -------
__global__ __launch_bounds__(256) void k_scores(const unsigned short* __restrict__ q,
                                                const unsigned short* __restrict__ kmat,
                                                float* __restrict__ sc) {
  const int lane = threadIdx.x & 31;
  int wave = blockIdx.x * (blockDim.x >> 5) + (threadIdx.x >> 5);
  const int nt = CT / 16;
  int kt = wave % nt; wave /= nt;
  int qt = wave % nt; wave /= nt;
  int bh = wave;
  if (bh >= CB * CH) return;
  if (kt > qt) return;  // fully masked tile: never read by softmax
  const int half = lane >> 4, l16 = lane & 15;
  const __bf16* arow = (const __bf16*)q + ((size_t)bh * CT + qt * 16 + l16) * CD;
  const __bf16* brow = (const __bf16*)kmat + ((size_t)bh * CT + kt * 16 + l16) * CD;
  v8f acc = zero8();
#pragma unroll
  for (int k0 = 0; k0 < CD; k0 += 32) {
    FragBF af, bf;
    af.h[0] = *(const v8bf*)(arow + k0 + 8 * half);
    af.h[1] = *(const v8bf*)(arow + k0 + 16 + 8 * half);
    bf.h[0] = *(const v8bf*)(brow + k0 + 8 * half);
    bf.h[1] = *(const v8bf*)(brow + k0 + 16 + 8 * half);
    acc = __builtin_amdgcn_wmma_f32_16x16x32_bf16(false, af.v, false, bf.v, (short)0,
                                                  acc, false, false);
  }
  float* base = sc + (size_t)bh * CT * CT;
  int c = kt * 16 + l16;
#pragma unroll
  for (int r = 0; r < 8; ++r) {
    int qr = qt * 16 + r + 8 * half;
    float v = acc[r] * CSCALE;
    if (c > qr) v = -__builtin_inff();
    base[(size_t)qr * CT + c] = v;
  }
}

// ---------------- causal softmax, zero-pad to 16-aligned tile edge -----------------
__global__ __launch_bounds__(256) void k_softmax(float* __restrict__ sc) {
  __shared__ float red[256];
  int row = blockIdx.x;  // over B*H*T
  int t = row % CT;
  int lim = ((t >> 4) + 1) << 4;
  float* p = sc + (size_t)row * CT;
  float m = -3.4e38f;
  for (int e = threadIdx.x; e <= t; e += 256) m = fmaxf(m, p[e]);
  red[threadIdx.x] = m;
  __syncthreads();
  for (int s = 128; s > 0; s >>= 1) {
    if ((int)threadIdx.x < s) red[threadIdx.x] = fmaxf(red[threadIdx.x], red[threadIdx.x + s]);
    __syncthreads();
  }
  m = red[0];
  __syncthreads();
  float sum = 0.f;
  for (int e = threadIdx.x; e <= t; e += 256) { float v = expf(p[e] - m); p[e] = v; sum += v; }
  red[threadIdx.x] = sum;
  __syncthreads();
  for (int s = 128; s > 0; s >>= 1) {
    if ((int)threadIdx.x < s) red[threadIdx.x] += red[threadIdx.x + s];
    __syncthreads();
  }
  float inv = 1.0f / red[0];
  for (int e = threadIdx.x; e <= t; e += 256) p[e] *= inv;
  for (int e = t + 1 + (int)threadIdx.x; e < lim; e += 256) p[e] = 0.0f;  // pad tile
}

// ---------------- O = P V : wave per (b,h,qtile,dchunk); K-loop stops at causal lim
__global__ __launch_bounds__(256) void k_attnout(const float* __restrict__ pr,
                                                 const unsigned short* __restrict__ vt,
                                                 float* __restrict__ ctx) {
  const int lane = threadIdx.x & 31;
  int wave = blockIdx.x * (blockDim.x >> 5) + (threadIdx.x >> 5);
  int dc = wave & 3; wave >>= 2;
  int qt = wave % (CT / 16); wave /= (CT / 16);
  int bh = wave;
  if (bh >= CB * CH) return;
  const int half = lane >> 4, l16 = lane & 15;
  int b = bh / CH, h = bh % CH;
  int m = qt * 16 + l16;
  const float* prow = pr + ((size_t)bh * CT + m) * CT;
  const __bf16* brow = (const __bf16*)vt + ((size_t)bh * CD + dc * 16 + l16) * CT;
  int lim = (qt + 1) * 16;
  v8f acc = zero8();
  for (int k0 = 0; k0 < lim; k0 += 32) {
    FragBF af, bf;
#pragma unroll
    for (int cc = 0; cc < 2; ++cc) {
      int base = k0 + 16 * cc + 8 * half;
      if (base < lim) {
        v8f pv = *(const v8f*)(prow + base);
#pragma unroll
        for (int j = 0; j < 8; ++j) af.s[cc * 8 + j] = f2bf(pv[j]);
      } else {
#pragma unroll
        for (int i = 0; i < 4; ++i) af.u[cc * 4 + i] = 0u;
      }
    }
    bf.h[0] = *(const v8bf*)(brow + k0 + 8 * half);
    bf.h[1] = *(const v8bf*)(brow + k0 + 16 + 8 * half);
    acc = __builtin_amdgcn_wmma_f32_16x16x32_bf16(false, af.v, false, bf.v, (short)0,
                                                  acc, false, false);
  }
#pragma unroll
  for (int r = 0; r < 8; ++r) {
    int qr = qt * 16 + r + 8 * half;
    int d = dc * 16 + l16;
    ctx[((size_t)b * CT + qr) * CE + h * CD + d] = acc[r];
  }
}

// ---------------- host-side orchestration ----------------
extern "C" void kernel_launch(void* const* d_in, const int* in_sizes, int n_in,
                              void* d_out, int out_size, void* d_ws, size_t ws_size,
                              hipStream_t stream) {
  (void)in_sizes; (void)n_in; (void)out_size; (void)ws_size;
  const int* tok = (const int*)d_in[0];
  const float* wte = (const float*)d_in[1];
  const float* wpe = (const float*)d_in[2];
  const float* lnf_g = (const float*)d_in[3];
  const float* head_w = (const float*)d_in[4];

  char* ws = (char*)d_ws;
  size_t off = 0;
  auto alloc = [&](size_t bytes) -> void* {
    void* p = ws + off;
    off += (bytes + 255) & ~(size_t)255;
    return p;
  };
  // peak workspace ~251 MB
  float* x = (float*)alloc((size_t)CB * CT * CE * 4);
  unsigned short* xn = (unsigned short*)alloc((size_t)CB * CT * CE * 2);
  float* qkv = (float*)alloc((size_t)CB * CT * 3 * CE * 4);
  unsigned short* qb = (unsigned short*)alloc((size_t)CB * CH * CT * CD * 2);
  unsigned short* kb = (unsigned short*)alloc((size_t)CB * CH * CT * CD * 2);
  unsigned short* vt = (unsigned short*)alloc((size_t)CB * CH * CD * CT * 2);
  float* sc = (float*)alloc((size_t)CB * CH * CT * CT * 4);
  float* ctx = (float*)alloc((size_t)CB * CT * CE * 4);
  unsigned short* cbf = (unsigned short*)alloc((size_t)CB * CT * CE * 2);
  float* hid = (float*)alloc((size_t)CB * CT * 4 * CE * 4);
  unsigned short* hbf = (unsigned short*)alloc((size_t)CB * CT * 4 * CE * 2);
  unsigned short* wt = (unsigned short*)alloc((size_t)CV * CE * 2);  // max transposed weight

  auto gemm = [&](const unsigned short* A, const unsigned short* Bt, const float* bias,
                  const float* resid, float* C, int M, int N, int K, int act) {
    int nsuper = (N + 511) >> 9;
    int blocks = (M / 32) * nsuper;
    k_gemm<<<blocks, 256, 0, stream>>>(A, Bt, bias, resid, C, M, N, K, act);
  };
  auto transw = [&](const float* W, int K, int N) {
    size_t n = (size_t)K * N;
    k_transw<<<(int)((n + 255) / 256), 256, 0, stream>>>(W, wt, K, N);
  };

  const int nBTE = CB * CT * CE;
  k_embed<<<(nBTE + 255) / 256, 256, 0, stream>>>(tok, wte, wpe, x);

  for (int l = 0; l < 12; ++l) {
    const float* ln1_g  = (const float*)d_in[5 + l * 10 + 0];
    const float* attn_w = (const float*)d_in[5 + l * 10 + 1];
    const float* attn_b = (const float*)d_in[5 + l * 10 + 2];
    const float* proj_w = (const float*)d_in[5 + l * 10 + 3];
    const float* proj_b = (const float*)d_in[5 + l * 10 + 4];
    const float* ln2_g  = (const float*)d_in[5 + l * 10 + 5];
    const float* fc_w   = (const float*)d_in[5 + l * 10 + 6];
    const float* fc_b   = (const float*)d_in[5 + l * 10 + 7];
    const float* fc2_w  = (const float*)d_in[5 + l * 10 + 8];
    const float* fc2_b  = (const float*)d_in[5 + l * 10 + 9];

    // attention
    k_ln<<<CB * CT, 256, 0, stream>>>(x, ln1_g, xn);
    transw(attn_w, CE, 3 * CE);
    gemm(xn, wt, attn_b, nullptr, qkv, CB * CT, 3 * CE, CE, 0);
    k_split<<<(nBTE + 255) / 256, 256, 0, stream>>>(qkv, qb, kb, vt);
    {
      int waves = CB * CH * (CT / 16) * (CT / 16);
      k_scores<<<(waves + 7) / 8, 256, 0, stream>>>(qb, kb, sc);
    }
    k_softmax<<<CB * CH * CT, 256, 0, stream>>>(sc);
    {
      int waves = CB * CH * (CT / 16) * (CD / 16);
      k_attnout<<<(waves + 7) / 8, 256, 0, stream>>>(sc, vt, ctx);
    }
    k_cvt<<<(nBTE + 255) / 256, 256, 0, stream>>>(ctx, cbf, nBTE);
    transw(proj_w, CE, CE);
    gemm(cbf, wt, proj_b, x, x, CB * CT, CE, CE, 0);  // fused residual into x

    // MLP
    k_ln<<<CB * CT, 256, 0, stream>>>(x, ln2_g, xn);
    transw(fc_w, CE, 4 * CE);
    gemm(xn, wt, fc_b, nullptr, hid, CB * CT, 4 * CE, CE, 1);  // fused exact GELU
    int nH = CB * CT * 4 * CE;
    k_cvt<<<(nH + 255) / 256, 256, 0, stream>>>(hid, hbf, nH);
    transw(fc2_w, 4 * CE, CE);
    gemm(hbf, wt, fc2_b, x, x, CB * CT, CE, 4 * CE, 0);  // fused residual into x
  }

  // final LN + LM head straight into d_out (fp32 logits)
  k_ln<<<CB * CT, 256, 0, stream>>>(x, lnf_g, xn);
  transw(head_w, CE, CV);
  gemm(xn, wt, nullptr, nullptr, (float*)d_out, CB * CT, CV, CE, 0);
}
